// sn_attention_8057358648239
// MI455X (gfx1250) — matrix operand order repaired and checked by reading, not compiled, not run on hardware
//
#include <hip/hip_runtime.h>
#include <stdint.h>

// Problem constants (from reference: B=4, C=128, H=W=D=24)
#define BN   4
#define CN   128
#define C8N  16
#define C2N  64
#define LN   13824      // 24^3
#define MN   1728       // 12^3
#define LT   864        // LN/16 query tiles per batch

typedef __attribute__((ext_vector_type(16))) __bf16   v16bf;
typedef __attribute__((ext_vector_type(8)))  float    v8f;
typedef __attribute__((ext_vector_type(4)))  uint32_t v4u;

union ABOp {
  v16bf    v;
  v4u      q[2];
  uint16_t h[16];
  uint32_t w[8];
};

static __device__ __forceinline__ uint16_t f2bf(float f) {
  uint32_t u = __float_as_uint(f);
  u += 0x7fffu + ((u >> 16) & 1u);   // round-to-nearest-even
  return (uint16_t)(u >> 16);
}

static __device__ __forceinline__ v8f wmma_bf16(const ABOp& a, const ABOp& b, v8f c) {
  // (neg_a, A, neg_b, B, c_mod, C, reuse_a, reuse_b)
  return __builtin_amdgcn_wmma_f32_16x16x32_bf16(false, a.v, false, b.v, (short)0, c, false, false);
}

// ---------------- Stage 1: theta = Wt @ x  -> bf16 [B][L][16] row-major ----------------
__global__ __launch_bounds__(256) void k_theta(const float* __restrict__ x,
                                               const float* __restrict__ Wt,
                                               const float* __restrict__ bt,
                                               uint16_t* __restrict__ theta) {
  __shared__ float w[C8N * CN];
  for (int i = threadIdx.x; i < C8N * CN; i += 256) w[i] = Wt[i];
  __syncthreads();
  int t = blockIdx.x * 256 + threadIdx.x;   // grid covers BN*LN exactly
  int b = t / LN, l = t % LN;
  float acc[C8N];
  #pragma unroll
  for (int o = 0; o < C8N; ++o) acc[o] = bt[o];
  const float* xp = x + (size_t)b * CN * LN + l;
  for (int c = 0; c < CN; ++c) {
    float xv = xp[(size_t)c * LN];
    #pragma unroll
    for (int o = 0; o < C8N; ++o) acc[o] = fmaf(w[o * CN + c], xv, acc[o]);
  }
  uint32_t* dst = (uint32_t*)(theta + ((size_t)b * LN + l) * C8N);
  #pragma unroll
  for (int o = 0; o < C8N; o += 2)
    dst[o >> 1] = (uint32_t)f2bf(acc[o]) | ((uint32_t)f2bf(acc[o + 1]) << 16);
}

// ------- Stage 1: phi = maxpool2(Wp @ x) -> bf16 [B][M][16] row-major (B operand) -------
__global__ __launch_bounds__(256) void k_phi(const float* __restrict__ x,
                                             const float* __restrict__ Wp,
                                             const float* __restrict__ bp,
                                             uint16_t* __restrict__ phi) {
  __shared__ float w[C8N * CN];
  for (int i = threadIdx.x; i < C8N * CN; i += 256) w[i] = Wp[i];
  __syncthreads();
  int t = blockIdx.x * 256 + threadIdx.x;   // grid covers BN*MN exactly
  int b = t / MN, m = t % MN;
  int dz = m % 12, wy = (m / 12) % 12, hx = m / 144;
  float res[C8N];
  #pragma unroll
  for (int o = 0; o < C8N; ++o) res[o] = -__builtin_inff();
  const float* xb = x + (size_t)b * CN * LN;
  for (int p = 0; p < 8; ++p) {
    int l = (2 * hx + ((p >> 2) & 1)) * 576 + (2 * wy + ((p >> 1) & 1)) * 24 + (2 * dz + (p & 1));
    float acc[C8N];
    #pragma unroll
    for (int o = 0; o < C8N; ++o) acc[o] = 0.f;
    for (int c = 0; c < CN; ++c) {
      float xv = xb[(size_t)c * LN + l];
      #pragma unroll
      for (int o = 0; o < C8N; ++o) acc[o] = fmaf(w[o * CN + c], xv, acc[o]);
    }
    #pragma unroll
    for (int o = 0; o < C8N; ++o) res[o] = fmaxf(res[o], acc[o]);
  }
  uint32_t* dst = (uint32_t*)(phi + ((size_t)b * MN + m) * C8N);
  #pragma unroll
  for (int o = 0; o < C8N; o += 2)
    dst[o >> 1] = (uint32_t)f2bf(res[o] + bp[o]) | ((uint32_t)f2bf(res[o + 1] + bp[o + 1]) << 16);
}

// --- Stage 1: gT = transpose(maxpool2(Wg @ x)) -> bf16 [B][64][M] (PV B operand layout) ---
__global__ __launch_bounds__(256) void k_g(const float* __restrict__ x,
                                           const float* __restrict__ Wg,
                                           const float* __restrict__ bg,
                                           uint16_t* __restrict__ gT) {
  __shared__ float w[16 * CN];
  int grp = blockIdx.x / 27;                 // 4 groups of 16 output channels; 27 blocks/group
  for (int i = threadIdx.x; i < 16 * CN; i += 256) w[i] = Wg[grp * 16 * CN + i];
  __syncthreads();
  int t = (blockIdx.x % 27) * 256 + threadIdx.x;
  int b = t / MN, m = t % MN;
  int dz = m % 12, wy = (m / 12) % 12, hx = m / 144;
  float res[16];
  #pragma unroll
  for (int o = 0; o < 16; ++o) res[o] = -__builtin_inff();
  const float* xb = x + (size_t)b * CN * LN;
  for (int p = 0; p < 8; ++p) {
    int l = (2 * hx + ((p >> 2) & 1)) * 576 + (2 * wy + ((p >> 1) & 1)) * 24 + (2 * dz + (p & 1));
    float acc[16];
    #pragma unroll
    for (int o = 0; o < 16; ++o) acc[o] = 0.f;
    for (int c = 0; c < CN; ++c) {
      float xv = xb[(size_t)c * LN + l];
      #pragma unroll
      for (int o = 0; o < 16; ++o) acc[o] = fmaf(w[o * CN + c], xv, acc[o]);
    }
    #pragma unroll
    for (int o = 0; o < 16; ++o) res[o] = fmaxf(res[o], acc[o]);
  }
  #pragma unroll
  for (int o = 0; o < 16; ++o)
    gT[((size_t)b * C2N + grp * 16 + o) * MN + m] = f2bf(res[o] + bg[grp * 16 + o]);
}

// ------------- Stage 1: pack Wc [128][64] f32 -> bf16, zero the operand pad block -------------
__global__ void k_wc(const float* __restrict__ Wc, uint16_t* __restrict__ Wcb,
                     uint16_t* __restrict__ zpad) {
  int i = blockIdx.x * 256 + threadIdx.x;
  if (i < CN * C2N) Wcb[i] = f2bf(Wc[i]);
  if (i < 32) zpad[i] = 0;   // 64-byte zero block: QK B-operand K=16..31 padding source
}

// ---------- Stage 2: fused attention (2-pass safe softmax) + final conv + residual ----------
// One wave per 16-query tile. 4 waves / block (independent tiles).
__global__ __launch_bounds__(128) void k_attn(const float* __restrict__ x,
                                              const uint16_t* __restrict__ theta,
                                              const uint16_t* __restrict__ phi,
                                              const uint16_t* __restrict__ gT,
                                              const uint16_t* __restrict__ Wcb,
                                              const uint16_t* __restrict__ zpad,
                                              const float* __restrict__ bc,
                                              const float* __restrict__ sigp,
                                              float* __restrict__ out) {
  __shared__ __align__(16) uint16_t Pbuf[4][16 * 32];   // P tile staging (C/D -> A layout)
  __shared__ __align__(16) uint16_t Obuf[4][16 * 64];   // O tile staging for final conv
  __shared__ float Ybuf[4][16 * 132];                   // padded (bank-conflict-free) output

  const int wid  = threadIdx.x >> 5;
  const int lane = threadIdx.x & 31;
  const int half = lane >> 4;
  const int r    = lane & 15;

  const int tile = blockIdx.x * 4 + wid;
  const int b  = tile / LT;
  const int l0 = (tile % LT) * 16;

  const v4u z4 = {0u, 0u, 0u, 0u};

  // theta A operand [16 x 32]: lanes<16 carry K0..15 (ch0..15);
  // lanes>=16 (K16..31) are zero padding. Lives in registers for both passes.
  ABOp aTh;
  {
    const uint16_t* tp = theta + ((size_t)b * LN + l0) * C8N;
    aTh.q[0] = half ? z4 : *(const v4u*)(tp + r * C8N);
    aTh.q[1] = half ? z4 : *(const v4u*)(tp + r * C8N + 8);
  }

  // Branch-free QK B-operand: lanes 0-15 stream phi rows (stride 1 KB/chunk);
  // lanes 16-31 (K=16..31 zero pad) read a pinned 64-B zero block with stride 0.
  const uint16_t* phiB = phi + (size_t)b * MN * C8N;
  const uint16_t* p0base = (half == 0) ? phiB + (size_t)r * C8N        : zpad;
  const uint16_t* p1base = (half == 0) ? phiB + (size_t)(16 + r) * C8N : zpad;
  const size_t    pstep  = (half == 0) ? (size_t)32 * C8N : 0;

  const uint16_t* gB = gT + (size_t)b * C2N * MN;
  uint16_t* pb = Pbuf[wid];
  v8f zf = {};

  // ================= Pass 1: exact row max, no cross-lane ops in the loop =================
  float mrow[8];
  #pragma unroll
  for (int i = 0; i < 8; ++i) mrow[i] = -__builtin_inff();

  {
    const uint16_t* pp0 = p0base;
    const uint16_t* pp1 = p1base;
    for (int m0 = 0; m0 < MN; m0 += 32) {
      ABOp b0, b1;
      b0.q[0] = *(const v4u*)pp0;  b0.q[1] = *(const v4u*)(pp0 + 8);
      b1.q[0] = *(const v4u*)pp1;  b1.q[1] = *(const v4u*)(pp1 + 8);
      pp0 += pstep;  pp1 += pstep;
      v8f s0 = wmma_bf16(aTh, b0, zf);
      v8f s1 = wmma_bf16(aTh, b1, zf);
      #pragma unroll
      for (int i = 0; i < 8; ++i) mrow[i] = fmaxf(mrow[i], fmaxf(s0[i], s1[i]));
    }
  }
  // one cross-lane reduction: rows live per half (row = vgpr index i + 8*half)
  #pragma unroll
  for (int d = 1; d < 16; d <<= 1) {
    #pragma unroll
    for (int i = 0; i < 8; ++i) mrow[i] = fmaxf(mrow[i], __shfl_xor(mrow[i], d, 32));
  }

  // ============ Pass 2: P = exp(S - m), O += P @ g, per-lane partial sums ============
  v8f acc[4];
  #pragma unroll
  for (int t = 0; t < 4; ++t) acc[t] = zf;
  float lsum[8];
  #pragma unroll
  for (int i = 0; i < 8; ++i) lsum[i] = 0.f;

  {
    const uint16_t* pp0 = p0base;
    const uint16_t* pp1 = p1base;
    for (int m0 = 0; m0 < MN; m0 += 32) {
      ABOp b0, b1;
      b0.q[0] = *(const v4u*)pp0;  b0.q[1] = *(const v4u*)(pp0 + 8);
      b1.q[0] = *(const v4u*)pp1;  b1.q[1] = *(const v4u*)(pp1 + 8);
      pp0 += pstep;  pp1 += pstep;
      v8f s0 = wmma_bf16(aTh, b0, zf);
      v8f s1 = wmma_bf16(aTh, b1, zf);

      float p0[8], p1[8];
      #pragma unroll
      for (int i = 0; i < 8; ++i) {
        p0[i] = __expf(s0[i] - mrow[i]);
        p1[i] = __expf(s1[i] - mrow[i]);
        lsum[i] += p0[i] + p1[i];
      }

      // P (C/D layout) -> LDS row-major -> reload in A layout
      #pragma unroll
      for (int i = 0; i < 8; ++i) {
        pb[(i + 8 * half) * 32 + r]      = f2bf(p0[i]);
        pb[(i + 8 * half) * 32 + 16 + r] = f2bf(p1[i]);
      }
      asm volatile("s_wait_dscnt 0" ::: "memory");
      ABOp aP;
      aP.q[0] = *(const v4u*)(pb + r * 32 + half * 8);
      aP.q[1] = *(const v4u*)(pb + r * 32 + 16 + half * 8);

      // O += P @ g  (K = 32 keys, 4 output-channel tiles of 16)
      #pragma unroll
      for (int t = 0; t < 4; ++t) {
        ABOp bg;  // gT[b][c2][m]: lane gets rows m0 + 8*half + 0..7 and m0 + 16 + 8*half + 0..7, col c2 = t*16+r
        const uint16_t* gp = gB + (size_t)(t * 16 + r) * MN + m0;
        bg.q[0] = *(const v4u*)(gp + half * 8);
        bg.q[1] = *(const v4u*)(gp + 16 + half * 8);
        acc[t] = wmma_bf16(aP, bg, acc[t]);
      }
    }
  }

  // one cross-lane reduction for the row sums, then normalize
  #pragma unroll
  for (int d = 1; d < 16; d <<= 1) {
    #pragma unroll
    for (int i = 0; i < 8; ++i) lsum[i] += __shfl_xor(lsum[i], d, 32);
  }
  #pragma unroll
  for (int i = 0; i < 8; ++i) {
    float inv = 1.f / lsum[i];
    #pragma unroll
    for (int t = 0; t < 4; ++t) acc[t][i] *= inv;
  }

  // ---- final conv: Y[16 x 128] = O[16 x 64] @ Wc^T, via LDS re-layout ----
  uint16_t* ob = Obuf[wid];
  #pragma unroll
  for (int t = 0; t < 4; ++t) {
    #pragma unroll
    for (int i = 0; i < 8; ++i)
      ob[(i + 8 * half) * 64 + t * 16 + r] = f2bf(acc[t][i]);
  }
  asm volatile("s_wait_dscnt 0" ::: "memory");

  ABOp aO0, aO1;  // K chunks 0..31 and 32..63 of O in A layout
  aO0.q[0] = *(const v4u*)(ob + r * 64 + half * 8);
  aO0.q[1] = *(const v4u*)(ob + r * 64 + 16 + half * 8);
  aO1.q[0] = *(const v4u*)(ob + r * 64 + 32 + half * 8);
  aO1.q[1] = *(const v4u*)(ob + r * 64 + 48 + half * 8);

  float* yb = Ybuf[wid];
  #pragma unroll
  for (int nt = 0; nt < 8; ++nt) {
    ABOp bw0, bw1;  // Wc[oc][c2] as B: K=c2, N=oc
    const uint16_t* wp = Wcb + (size_t)(nt * 16 + r) * C2N;
    bw0.q[0] = *(const v4u*)(wp + half * 8);
    bw0.q[1] = *(const v4u*)(wp + 16 + half * 8);
    bw1.q[0] = *(const v4u*)(wp + 32 + half * 8);
    bw1.q[1] = *(const v4u*)(wp + 48 + half * 8);
    v8f y = zf;
    y = wmma_bf16(aO0, bw0, y);
    y = wmma_bf16(aO1, bw1, y);
    float bcv = bc[nt * 16 + r];
    #pragma unroll
    for (int i = 0; i < 8; ++i)
      yb[(i + 8 * half) * 132 + nt * 16 + r] = y[i] + bcv;
  }
  asm volatile("s_wait_dscnt 0" ::: "memory");

  // ---- residual blend, coalesced along l ----
  const float sig = *sigp;
  const float rs  = 1.f / (1.f + sig);
  const float* xb = x   + (size_t)b * CN * LN;
  float* outb     = out + (size_t)b * CN * LN;
  for (int j = 0; j < 64; ++j) {
    int oc = j * 2 + half;
    size_t off = (size_t)oc * LN + (l0 + r);
    outb[off] = (xb[off] + sig * yb[r * 132 + oc]) * rs;
  }
}

extern "C" void kernel_launch(void* const* d_in, const int* in_sizes, int n_in,
                              void* d_out, int out_size, void* d_ws, size_t ws_size,
                              hipStream_t stream) {
  const float* x  = (const float*)d_in[0];
  const float* Wt = (const float*)d_in[1];
  const float* bt = (const float*)d_in[2];
  const float* Wp = (const float*)d_in[3];
  const float* bp = (const float*)d_in[4];
  const float* Wg = (const float*)d_in[5];
  const float* bg = (const float*)d_in[6];
  const float* Wc = (const float*)d_in[7];
  const float* bc = (const float*)d_in[8];
  const float* sg = (const float*)d_in[9];

  char* ws = (char*)d_ws;
  uint16_t* theta = (uint16_t*)(ws);             // 4*13824*16*2 = 1,769,472 B
  uint16_t* phi   = (uint16_t*)(ws + 1769472);   // 4*1728*16*2  =   221,184 B
  uint16_t* gT    = (uint16_t*)(ws + 1990656);   // 4*64*1728*2  =   884,736 B
  uint16_t* Wcb   = (uint16_t*)(ws + 2875392);   // 128*64*2     =    16,384 B
  uint16_t* zpad  = (uint16_t*)(ws + 2891776);   // 64 B zero block (total < 3 MB)

  k_theta<<<(BN * LN) / 256, 256, 0, stream>>>(x, Wt, bt, theta);
  k_phi<<<(BN * MN) / 256, 256, 0, stream>>>(x, Wp, bp, phi);
  k_g<<<4 * (BN * MN) / 256, 256, 0, stream>>>(x, Wg, bg, gT);
  k_wc<<<(CN * C2N + 255) / 256, 256, 0, stream>>>(Wc, Wcb, zpad);
  k_attn<<<(BN * LT) / 4, 128, 0, stream>>>(x, theta, phi, gT, Wcb, zpad, bc, sg, (float*)d_out);

  (void)in_sizes; (void)n_in; (void)out_size; (void)ws_size;
}
